// DeepMLPWithoutSkip_884763263792
// MI455X (gfx1250) — compile-verified, matrix operand order checked by reading, not run on hardware
//
#include <hip/hip_runtime.h>

typedef __attribute__((ext_vector_type(16))) _Float16 v16h;
typedef __attribute__((ext_vector_type(8)))  float    v8f;
typedef __attribute__((ext_vector_type(4)))  unsigned int v4u;

#define NB       524288
#define D_INDIM  10
#define HDIM     64
#define N_HID    98
#define LSTRIDE  8448   // per-layer blob: 8192B fp16 W^T + 256B f32 bias (16B aligned)

union ABfrag { v16h h; v4u u[2]; };
union BFfrag { v8f  f; v4u u[2]; };

__device__ __forceinline__ float sigm(float x) {
    // sigmoid(x) = 1 / (1 + 2^(-x*log2(e)))  -> v_exp_f32 + v_rcp_f32
    return __builtin_amdgcn_rcpf(1.0f + __builtin_amdgcn_exp2f(-1.44269504088896f * x));
}

// Pre-pass: build per-layer blobs in d_ws:
//   blob[l][0:8192)    = W^T fp16, row-major [out][in] (128B rows)
//   blob[l][8192:8448) = bias f32 [64]
__global__ void __launch_bounds__(256)
wt_pack_kernel(const float* __restrict__ W_hid, const float* __restrict__ b_hid,
               unsigned char* __restrict__ blob) {
    const int NW = N_HID * HDIM * HDIM;
    int idx = blockIdx.x * 256 + threadIdx.x;
    if (idx < NW) {
        int l = idx >> 12;
        int r = idx & 4095;
        int i = r >> 6;   // in-feature
        int o = r & 63;   // out-feature
        *(_Float16*)(blob + (long)l * LSTRIDE + (((o << 6) + i) << 1)) =
            (_Float16)W_hid[idx];
    } else if (idx < NW + N_HID * HDIM) {
        int j = idx - NW;
        int l = j >> 6;
        int o = j & 63;
        *(float*)(blob + (long)l * LSTRIDE + 8192 + (o << 2)) = b_hid[j];
    }
}

// Issue async copy of one layer blob into LDS buffer (whole block cooperates).
// Each lane moves 16B per instruction; 256 threads x 2 = 8KB weights, first 16
// threads move the 256B bias. Tracked by ASYNCcnt, no VGPR staging, no DScnt.
__device__ __forceinline__ void issue_async_layer(unsigned lds_base, int tid,
                                                  const unsigned char* gsrc) {
    unsigned       laddr = lds_base + (unsigned)(tid * 16);
    unsigned long long gaddr =
        (unsigned long long)(uintptr_t)(gsrc + (long)tid * 16);
    asm volatile("global_load_async_to_lds_b128 %0, %1, off"
                 :: "v"(laddr), "v"(gaddr) : "memory");
    asm volatile("global_load_async_to_lds_b128 %0, %1, off offset:4096"
                 :: "v"(laddr), "v"(gaddr) : "memory");
    if (tid < 16)
        asm volatile("global_load_async_to_lds_b128 %0, %1, off offset:8192"
                     :: "v"(laddr), "v"(gaddr) : "memory");
}

__device__ __forceinline__ void wait_async0() {
    asm volatile("s_wait_asynccnt 0x0" ::: "memory");
}

// Main kernel: 256 threads = 8 waves, 32 rows/wave, 256 rows/block.
// Activations live in registers (WMMA fragments) across all 98 layers.
__global__ void __launch_bounds__(256)
deep_mlp_kernel(const float* __restrict__ x,
                const float* __restrict__ W_in,  const float* __restrict__ b_in,
                const unsigned char* __restrict__ blob,
                const float* __restrict__ W_out, const float* __restrict__ b_out,
                float* __restrict__ out) {
    __shared__ __align__(16) unsigned char smraw[2][LSTRIDE];

    const int tid   = threadIdx.x;
    const int lane  = tid & 31;
    const int wave  = tid >> 5;
    const int lh    = lane >> 4;          // 0: lanes 0-15, 1: lanes 16-31
    const int lh8   = lh * 8;
    const int lh16  = lh * 16;
    const int lrow  = lane & 15;
    const int rowbase = blockIdx.x * 256 + wave * 32;

    const unsigned lds0 = (unsigned)(uintptr_t)&smraw[0][0];
    const unsigned lds1 = (unsigned)(uintptr_t)&smraw[1][0];

    // ---- stage W_in (640 f32) + b_in (64 f32) into buffer 0 ----
    {
        float* swf = (float*)&smraw[0][0];
        for (int i = tid; i < 704; i += 256)
            swf[i] = (i < 640) ? W_in[i] : b_in[i - 640];
    }
    // kick off async copy of layer 0 weights into buffer 1 (overlaps layer-0 VALU)
    issue_async_layer(lds1, tid, blob);
    __syncthreads();

    // ---- layer 0: h0 = sigmoid(x @ W_in + b_in) built directly as B-fragments ----
    // B-fragment slot s of chunk c holds feature K = 32c + (s>=8 ? 16 : 0) + lh8 + (s&7)
    v16h Bf[2][2];
    {
        const float* swf = (const float*)&smraw[0][0];
        for (int rt = 0; rt < 2; ++rt) {
            const long row = (long)rowbase + rt * 16 + lrow;
            const float* xr = x + row * D_INDIM;
            float xv[D_INDIM];
            #pragma unroll
            for (int k = 0; k < D_INDIM; ++k) xv[k] = xr[k];
            #pragma unroll
            for (int c = 0; c < 2; ++c) {
                #pragma unroll
                for (int s = 0; s < 16; ++s) {
                    const int K = c * 32 + ((s & 8) ? 16 : 0) + lh8 + (s & 7);
                    float a = swf[640 + K];
                    #pragma unroll
                    for (int k = 0; k < D_INDIM; ++k) a += xv[k] * swf[k * 64 + K];
                    Bf[rt][c][s] = (_Float16)sigm(a);
                }
            }
        }
    }

    // ---- 98 hidden layers, double-buffered async weight pipeline ----
    // layer l lives in buffer 1-(l&1); while computing l we async-fill buffer (l&1)
    for (int l = 0; l < N_HID; ++l) {
        wait_async0();       // own async loads for layer l complete
        __syncthreads();     // whole block's loads complete; prev buffer reads done
        if (l + 1 < N_HID)
            issue_async_layer((l & 1) ? lds1 : lds0, tid,
                              blob + (long)(l + 1) * LSTRIDE);

        const unsigned char* wb = (l & 1) ? &smraw[0][0] : &smraw[1][0];

        v8f acc[2][4];
        // K-chunk 0, bias fragment (from LDS) as the C operand
        #pragma unroll
        for (int m = 0; m < 4; ++m) {
            BFfrag bf;
            const unsigned char* bp = wb + 8192 + (m * 16 + lh8) * 4;
            bf.u[0] = *(const v4u*)bp;
            bf.u[1] = *(const v4u*)(bp + 16);
            ABfrag a;
            const int ro = (m * 16 + lrow) * 128 + lh16;
            a.u[0] = *(const v4u*)(wb + ro);
            a.u[1] = *(const v4u*)(wb + ro + 32);
            acc[0][m] = __builtin_amdgcn_wmma_f32_16x16x32_f16(
                false, a.h, false, Bf[0][0], (short)0, bf.f, false, false);
            acc[1][m] = __builtin_amdgcn_wmma_f32_16x16x32_f16(
                false, a.h, false, Bf[1][0], (short)0, bf.f, false, false);
        }
        // K-chunk 1, accumulate
        #pragma unroll
        for (int m = 0; m < 4; ++m) {
            ABfrag a;
            const int ro = (m * 16 + lrow) * 128 + 64 + lh16;
            a.u[0] = *(const v4u*)(wb + ro);
            a.u[1] = *(const v4u*)(wb + ro + 32);
            acc[0][m] = __builtin_amdgcn_wmma_f32_16x16x32_f16(
                false, a.h, false, Bf[0][1], (short)0, acc[0][m], false, false);
            acc[1][m] = __builtin_amdgcn_wmma_f32_16x16x32_f16(
                false, a.h, false, Bf[1][1], (short)0, acc[1][m], false, false);
        }
        // sigmoid + repack: D-tile layout == next layer's B-fragment layout (per-lane)
        #pragma unroll
        for (int rt = 0; rt < 2; ++rt) {
            v16h nb0, nb1;
            #pragma unroll
            for (int s = 0; s < 8; ++s) {
                nb0[s]     = (_Float16)sigm(acc[rt][0][s]);
                nb0[s + 8] = (_Float16)sigm(acc[rt][1][s]);
                nb1[s]     = (_Float16)sigm(acc[rt][2][s]);
                nb1[s + 8] = (_Float16)sigm(acc[rt][3][s]);
            }
            Bf[rt][0] = nb0;
            Bf[rt][1] = nb1;
        }
    }

    // ---- output layer: y = h @ W_out + b_out  (64 -> 2), cross-half reduce ----
    const float bo0 = b_out[0], bo1 = b_out[1];
    for (int rt = 0; rt < 2; ++rt) {
        float p0 = 0.0f, p1 = 0.0f;
        #pragma unroll
        for (int c = 0; c < 2; ++c) {
            #pragma unroll
            for (int s = 0; s < 16; ++s) {
                const int K = c * 32 + ((s & 8) ? 16 : 0) + lh8 + (s & 7);
                const float h = (float)Bf[rt][c][s];
                p0 += h * W_out[K * 2 + 0];
                p1 += h * W_out[K * 2 + 1];
            }
        }
        // each lane holds half the features of its row; partner is lane ^ 16 (SWAPX16)
        p0 += __int_as_float(__builtin_amdgcn_ds_swizzle(__float_as_int(p0), 0x401F));
        p1 += __int_as_float(__builtin_amdgcn_ds_swizzle(__float_as_int(p1), 0x401F));
        if (lane < 16) {
            const long row = (long)rowbase + rt * 16 + lrow;
            out[row * 2 + 0] = p0 + bo0;
            out[row * 2 + 1] = p1 + bo1;
        }
    }
}

extern "C" void kernel_launch(void* const* d_in, const int* in_sizes, int n_in,
                              void* d_out, int out_size, void* d_ws, size_t ws_size,
                              hipStream_t stream) {
    const float* x     = (const float*)d_in[0];
    const float* W_in  = (const float*)d_in[1];
    const float* b_in  = (const float*)d_in[2];
    const float* W_hid = (const float*)d_in[3];
    const float* b_hid = (const float*)d_in[4];
    const float* W_out = (const float*)d_in[5];
    const float* b_out = (const float*)d_in[6];
    float* out = (float*)d_out;

    unsigned char* blob = (unsigned char*)d_ws;  // 98 * 8448 B = 809 KB

    const int nElems = N_HID * HDIM * HDIM + N_HID * HDIM;
    wt_pack_kernel<<<(nElems + 255) / 256, 256, 0, stream>>>(W_hid, b_hid, blob);

    deep_mlp_kernel<<<NB / 256, 256, 0, stream>>>(x, W_in, b_in, blob,
                                                  W_out, b_out, out);
}
